// SRUpp_38139309588937
// MI455X (gfx1250) — matrix operand order repaired and checked by reading, not compiled, not run on hardware
//
#include <hip/hip_runtime.h>
#include <hip/hip_bf16.h>

typedef __bf16 bf16_t;
typedef __attribute__((ext_vector_type(16))) __bf16 v16bf;
typedef __attribute__((ext_vector_type(8)))  __bf16 v8bf;
typedef __attribute__((ext_vector_type(8)))  float  v8f;

#define BM 128
#define BN 64
#define BK 64
#define KP 8   // pad (bf16 elems): row stride (64+8)*2 = 144 B, 16B-aligned

// CDNA5 async memory->LDS copy, 16B per lane, no VGPR round-trip (ASYNCcnt).
__device__ __forceinline__ void async_copy_b128(bf16_t* lds_dst, const bf16_t* gsrc) {
    // LDS flat addresses carry the LDS byte offset in the low 32 bits.
    unsigned loff = (unsigned)(unsigned long long)lds_dst;
    asm volatile("global_load_async_to_lds_b128 %0, %1, off"
                 :: "v"(loff), "v"(gsrc)
                 : "memory");
}
__device__ __forceinline__ void wait_async() {
    asm volatile("s_wait_asynccnt 0x0" ::: "memory");
}

// ---------------- f32 -> bf16 conversion ----------------
__global__ __launch_bounds__(256) void cvt_f32_bf16(const float* __restrict__ src,
                                                    bf16_t* __restrict__ dst, long n) {
    long i = (long)blockIdx.x * blockDim.x + threadIdx.x;
    long stride = (long)gridDim.x * blockDim.x;
    for (; i < n; i += stride) dst[i] = (bf16_t)src[i];
}

// ---------------- generic bf16 WMMA GEMM ----------------
// C[M,N] = scale * (A[M,K] @ B)  (+ bias[n])  (+ C if acc)
// B row-major KxN (transB=0) or row-major NxK read as B^T (transB=1).
// Optional bf16 mirror output Cb. Batched via blockIdx.z (element strides sA/sB/sC).
__global__ __launch_bounds__(256) void gemm_bf16_wmma(
    const bf16_t* __restrict__ A, const bf16_t* __restrict__ B,
    const float* __restrict__ bias, float* __restrict__ C, bf16_t* __restrict__ Cb,
    int M, int N, int K, int lda, int ldb, int ldc,
    long sA, long sB, long sC, float scale, int transB, int acc)
{
    __shared__ alignas(16) bf16_t la [BM][BK + KP];   // row-major A tile
    __shared__ alignas(16) bf16_t lbT[BN][BK + KP];   // B tile stored [n][k]

    const int bz = blockIdx.z;
    A += (long)bz * sA;
    B += (long)bz * sB;
    const long coff = (long)bz * sC;

    const int m0   = blockIdx.y * BM;
    const int n0   = blockIdx.x * BN;
    const int tid  = threadIdx.x;
    const int lane = tid & 31;
    const int wave = tid >> 5;
    const int wm   = (wave & 3) * 32;   // 4 waves along M
    const int wn   = (wave >> 2) * 32;  // 2 waves along N
    const int l15  = lane & 15;
    const int lhi  = lane >> 4;

    v8f c[2][2] = {};

    for (int k0 = 0; k0 < K; k0 += BK) {
        __syncthreads();
        // ---- stage A: 128x64 bf16 = 1024 x (8-bf16 chunks); 4 async copies/thread ----
        #pragma unroll
        for (int j = 0; j < 4; ++j) {
            const int chunk = tid + j * 256;        // 0..1023
            const int r  = chunk >> 3;              // BK/8 = 8 chunks per row
            const int kc = (chunk & 7) << 3;
            async_copy_b128(&la[r][kc], &A[(long)(m0 + r) * lda + (k0 + kc)]);
        }
        // ---- stage B (transposed into lbT[n][k]): 64x64 = 512 chunks; 2/thread ----
        if (transB) {
            #pragma unroll
            for (int j = 0; j < 2; ++j) {
                const int chunk = tid + j * 256;    // 0..511
                const int r  = chunk >> 3;          // n row
                const int kc = (chunk & 7) << 3;
                async_copy_b128(&lbT[r][kc], &B[(long)(n0 + r) * ldb + (k0 + kc)]);
            }
        } else {
            // transpose scatter: load both chunks first, then scatter (batched waits)
            uint4 d[2];
            int rr[2], nn[2];
            #pragma unroll
            for (int j = 0; j < 2; ++j) {
                const int chunk = tid + j * 256;    // 0..511
                rr[j] = chunk >> 3;                 // k row (BN/8 = 8 chunks per row)
                nn[j] = (chunk & 7) << 3;
                d[j]  = *(const uint4*)&B[(long)(k0 + rr[j]) * ldb + (n0 + nn[j])];
            }
            #pragma unroll
            for (int j = 0; j < 2; ++j) {
                const bf16_t* e = (const bf16_t*)&d[j];
                #pragma unroll
                for (int q = 0; q < 8; ++q) lbT[nn[j] + q][rr[j]] = e[q];
            }
        }
        wait_async();
        __syncthreads();

        // ---- two WMMA K-steps per stage ----
        #pragma unroll
        for (int ks = 0; ks < 2; ++ks) {
            v16bf a[2], b[2];
            #pragma unroll
            for (int s = 0; s < 2; ++s) {
                // A frag: lane row m, two contiguous 8-elem K runs (CDNA5 A 16x32 layout)
                const int m  = wm + s * 16 + l15;
                const int ka = ks * 32 + lhi * 8;
                const v8bf alo = *(const v8bf*)&la[m][ka];
                const v8bf ahi = *(const v8bf*)&la[m][ka + 16];
                a[s] = __builtin_shufflevector(alo, ahi,
                        0,1,2,3,4,5,6,7,8,9,10,11,12,13,14,15);
                // B frag: lane col n, two contiguous 8-elem K runs (same K order as A)
                const int n  = wn + s * 16 + l15;
                const int kb = ks * 32 + lhi * 8;
                const v8bf blo = *(const v8bf*)&lbT[n][kb];
                const v8bf bhi = *(const v8bf*)&lbT[n][kb + 16];
                b[s] = __builtin_shufflevector(blo, bhi,
                        0,1,2,3,4,5,6,7,8,9,10,11,12,13,14,15);
            }
            #pragma unroll
            for (int i = 0; i < 2; ++i)
                #pragma unroll
                for (int j = 0; j < 2; ++j)
                    c[i][j] = __builtin_amdgcn_wmma_f32_16x16x32_bf16(
                        false, a[i], false, b[j], (short)0, c[i][j], false, false);
        }
    }

    // ---- store: C/D layout — VGPR v holds row (v + 8*lhi), col = l15 ----
    #pragma unroll
    for (int i = 0; i < 2; ++i) {
        #pragma unroll
        for (int j = 0; j < 2; ++j) {
            #pragma unroll
            for (int v = 0; v < 8; ++v) {
                const int m = m0 + wm + i * 16 + v + 8 * lhi;
                const int n = n0 + wn + j * 16 + l15;
                const long idx = coff + (long)m * ldc + n;
                float val = c[i][j][v] * scale;
                if (bias) val += bias[n];
                if (acc)  val += C[idx];
                if (C)  C[idx]  = val;
                if (Cb) Cb[idx] = (bf16_t)val;
            }
        }
    }
}

// ---------------- row softmax: f32 logits -> bf16 probs ----------------
__global__ __launch_bounds__(256) void softmax_rows(const float* __restrict__ logits,
                                                    bf16_t* __restrict__ probs, int Lrow) {
    const long row = blockIdx.x;
    const float* p = logits + row * (long)Lrow;
    bf16_t*      q = probs  + row * (long)Lrow;
    __shared__ float red[256];
    const int tid = threadIdx.x;

    float mx = -1e30f;
    for (int i = tid; i < Lrow; i += 256) mx = fmaxf(mx, p[i]);
    red[tid] = mx; __syncthreads();
    for (int s = 128; s > 0; s >>= 1) {
        if (tid < s) red[tid] = fmaxf(red[tid], red[tid + s]);
        __syncthreads();
    }
    mx = red[0]; __syncthreads();

    float sum = 0.f;
    for (int i = tid; i < Lrow; i += 256) sum += __expf(p[i] - mx);
    red[tid] = sum; __syncthreads();
    for (int s = 128; s > 0; s >>= 1) {
        if (tid < s) red[tid] += red[tid + s];
        __syncthreads();
    }
    const float inv = 1.f / red[0];
    for (int i = tid; i < Lrow; i += 256) q[i] = (bf16_t)(__expf(p[i] - mx) * inv);
}

// ---------------- Z = alpha*F + Q  (bf16 out) ----------------
__global__ __launch_bounds__(256) void z_combine(const float* __restrict__ F,
                                                 const float* __restrict__ Qf,
                                                 const float* __restrict__ alpha_p,
                                                 bf16_t* __restrict__ Zb, long n) {
    const float alpha = alpha_p[0];
    long i = (long)blockIdx.x * blockDim.x + threadIdx.x;
    long stride = (long)gridDim.x * blockDim.x;
    for (; i < n; i += stride) Zb[i] = (bf16_t)(alpha * F[i] + Qf[i]);
}

// ---------------- gated recurrence: 1 thread per (b,d) channel ----------------
__global__ __launch_bounds__(256) void sru_scan(
    const float* __restrict__ U, const float* __restrict__ x,
    const float* __restrict__ hinit,
    const float* __restrict__ v_f, const float* __restrict__ v_r,
    const float* __restrict__ bias_f, const float* __restrict__ bias_r,
    float* __restrict__ out, int Bn, int Ln, int Dn)
{
    const int g = blockIdx.x * blockDim.x + threadIdx.x;
    if (g >= Bn * Dn) return;
    const int b = g / Dn, d = g % Dn;
    float c = hinit[(long)b * Dn + d];
    const float vf = v_f[d], vr = v_r[d], bfv = bias_f[d], brv = bias_r[d];
    const long D3 = 3L * Dn;
    for (int t = 0; t < Ln; ++t) {
        const long ub = ((long)b * Ln + t) * D3;
        const float uf = U[ub + d];
        const float ur = U[ub + Dn + d];
        const float uh = U[ub + 2 * Dn + d];
        const float xv = x[((long)b * Ln + t) * Dn + d];
        const float f  = 1.f / (1.f + __expf(-(uf + vf * c + bfv)));
        const float cn = f * c + (1.f - f) * uh;
        const float r  = 1.f / (1.f + __expf(-(ur + vr * c + brv)));  // uses c_prev
        out[((long)b * Ln + t) * Dn + d] = r * cn + (1.f - r) * xv;
        c = cn;
    }
}

// ---------------- host launcher ----------------
extern "C" void kernel_launch(void* const* d_in, const int* in_sizes, int n_in,
                              void* d_out, int out_size, void* d_ws, size_t ws_size,
                              hipStream_t stream) {
    const int Bc = 4, Lc = 2048, Dc = 1024, Ac = 256;
    const int D3 = 3 * Dc;
    const long BL = (long)Bc * Lc;        // 8192

    const float* x      = (const float*)d_in[0];
    const float* hinit  = (const float*)d_in[1];
    const float* Wp     = (const float*)d_in[2];
    const float* bp     = (const float*)d_in[3];
    const float* v_f    = (const float*)d_in[4];
    const float* v_r    = (const float*)d_in[5];
    const float* bias_f = (const float*)d_in[6];
    const float* bias_r = (const float*)d_in[7];
    const float* Wq     = (const float*)d_in[8];
    const float* Wk     = (const float*)d_in[9];
    const float* Wv     = (const float*)d_in[10];
    const float* Wu     = (const float*)d_in[11];
    const float* alpha  = (const float*)d_in[12];
    float* out = (float*)d_out;

    // workspace carve-out (256B aligned)
    char* base = (char*)d_ws;
    size_t off = 0;
    auto carve = [&](size_t bytes) -> void* {
        off = (off + 255) & ~(size_t)255;
        void* p = base + off;
        off += bytes;
        return p;
    };
    bf16_t* xb   = (bf16_t*)carve(BL * Dc * 2);             // 16 MB
    bf16_t* Wpb  = (bf16_t*)carve((size_t)Dc * D3 * 2);     // 6 MB
    bf16_t* Wqb  = (bf16_t*)carve((size_t)Dc * Ac * 2);
    bf16_t* Wkb  = (bf16_t*)carve((size_t)Dc * Ac * 2);
    bf16_t* Wvb  = (bf16_t*)carve((size_t)Dc * Ac * 2);
    bf16_t* Wub  = (bf16_t*)carve((size_t)Ac * D3 * 2);
    float*  U    = (float*)carve(BL * D3 * 4);              // 96 MB
    float*  Qf   = (float*)carve(BL * Ac * 4);              // 8 MB
    bf16_t* Qb   = (bf16_t*)carve(BL * Ac * 2);
    bf16_t* Kb   = (bf16_t*)carve(BL * Ac * 2);
    bf16_t* Vb   = (bf16_t*)carve(BL * Ac * 2);
    float*  Lg   = (float*)carve((size_t)Bc * Lc * Lc * 4); // 64 MB logits
    bf16_t* Pb   = (bf16_t*)carve((size_t)Bc * Lc * Lc * 2);// 32 MB probs
    float*  F    = (float*)carve(BL * Ac * 4);              // 8 MB
    bf16_t* Zb   = (bf16_t*)carve(BL * Ac * 2);
    (void)ws_size;

    // 1) convert activations & weights to bf16
    cvt_f32_bf16<<<2048, 256, 0, stream>>>(x,  xb,  BL * Dc);
    cvt_f32_bf16<<<1024, 256, 0, stream>>>(Wp, Wpb, (long)Dc * D3);
    cvt_f32_bf16<<<256,  256, 0, stream>>>(Wq, Wqb, (long)Dc * Ac);
    cvt_f32_bf16<<<256,  256, 0, stream>>>(Wk, Wkb, (long)Dc * Ac);
    cvt_f32_bf16<<<256,  256, 0, stream>>>(Wv, Wvb, (long)Dc * Ac);
    cvt_f32_bf16<<<256,  256, 0, stream>>>(Wu, Wub, (long)Ac * D3);

    // 2) U_lin = x@Wp + bp  (8192x3072, K=1024)
    gemm_bf16_wmma<<<dim3(D3 / BN, BL / BM, 1), 256, 0, stream>>>(
        xb, Wpb, bp, U, nullptr, (int)BL, D3, Dc, Dc, D3, D3,
        0, 0, 0, 1.0f, 0, 0);

    // 3) Q (f32 + bf16), K (bf16), V (bf16): 8192x256, K=1024
    gemm_bf16_wmma<<<dim3(Ac / BN, BL / BM, 1), 256, 0, stream>>>(
        xb, Wqb, nullptr, Qf, Qb, (int)BL, Ac, Dc, Dc, Ac, Ac,
        0, 0, 0, 1.0f, 0, 0);
    gemm_bf16_wmma<<<dim3(Ac / BN, BL / BM, 1), 256, 0, stream>>>(
        xb, Wkb, nullptr, nullptr, Kb, (int)BL, Ac, Dc, Dc, Ac, Ac,
        0, 0, 0, 1.0f, 0, 0);
    gemm_bf16_wmma<<<dim3(Ac / BN, BL / BM, 1), 256, 0, stream>>>(
        xb, Wvb, nullptr, nullptr, Vb, (int)BL, Ac, Dc, Dc, Ac, Ac,
        0, 0, 0, 1.0f, 0, 0);

    // 4) logits = Q @ K^T / sqrt(A), batched over B  (2048x2048, K=256)
    gemm_bf16_wmma<<<dim3(Lc / BN, Lc / BM, Bc), 256, 0, stream>>>(
        Qb, Kb, nullptr, Lg, nullptr, Lc, Lc, Ac, Ac, Ac, Lc,
        (long)Lc * Ac, (long)Lc * Ac, (long)Lc * Lc,
        1.0f / 16.0f /* 1/sqrt(256) */, 1, 0);

    // 5) softmax rows -> bf16 probs
    softmax_rows<<<(int)(BL), 256, 0, stream>>>(Lg, Pb, Lc);

    // 6) F = P @ V, batched  (2048x256, K=2048)
    gemm_bf16_wmma<<<dim3(Ac / BN, Lc / BM, Bc), 256, 0, stream>>>(
        Pb, Vb, nullptr, F, nullptr, Lc, Ac, Lc, Lc, Ac, Ac,
        (long)Lc * Lc, (long)Lc * Ac, (long)Lc * Ac, 1.0f, 0, 0);

    // 7) Z = alpha*F + Q  (bf16)
    z_combine<<<2048, 256, 0, stream>>>(F, Qf, alpha, Zb, BL * Ac);

    // 8) U += Z @ Wu  (8192x3072, K=256, accumulate)
    gemm_bf16_wmma<<<dim3(D3 / BN, BL / BM, 1), 256, 0, stream>>>(
        Zb, Wub, nullptr, U, nullptr, (int)BL, D3, Ac, Ac, D3, D3,
        0, 0, 0, 1.0f, 0, 1);

    // 9) gated recurrence over time; one thread per (b,d) channel
    sru_scan<<<(Bc * Dc + 255) / 256, 256, 0, stream>>>(
        U, x, hinit, v_f, v_r, bias_f, bias_r, out, Bc, Lc, Dc);
}
